// Node_emblayer_33809982554710
// MI455X (gfx1250) — matrix operand order repaired and checked by reading, not compiled, not run on hardware
//
#include <hip/hip_runtime.h>
#include <stdint.h>

#define N_NODES 8192
#define N_EDGES 32768
#define DIM     512
#define DEG     16
#define DEP     8
#define HOPS    3

typedef __attribute__((ext_vector_type(16))) __bf16    v16bf;
typedef __attribute__((ext_vector_type(8)))  float     v8f;
typedef __attribute__((ext_vector_type(4)))  uint32_t  u32x4;

#define AS1 __attribute__((address_space(1)))
typedef const AS1 uint16_t* gcu16;
typedef const AS1 uint32_t* gcu32;
typedef const AS1 u32x4*    gcv4;
typedef const AS1 float*    gcf32;
typedef AS1 uint16_t*       gu16;
typedef AS1 uint32_t*       gu32;
typedef AS1 float*          gf32;

// ---------- bf16 helpers (RNE) ----------
__device__ __forceinline__ float bf2f(uint16_t h) {
    union { uint32_t u; float f; } x; x.u = ((uint32_t)h) << 16; return x.f;
}
__device__ __forceinline__ uint16_t f2bf(float f) {
    union { float f; uint32_t u; } x; x.f = f;
    uint32_t u = x.u;
    uint32_t r = u + 0x7FFFu + ((u >> 16) & 1u);
    return (uint16_t)(r >> 16);
}
__device__ __forceinline__ uint32_t pack2(float a, float b) {
    return (uint32_t)f2bf(a) | ((uint32_t)f2bf(b) << 16);
}

// ---------- f32 -> bf16 convert (contiguous, pairs) ----------
__global__ void conv_bf16_kernel(const float* __restrict__ in,
                                 uint16_t* __restrict__ out, int n2) {
    int i = blockIdx.x * blockDim.x + threadIdx.x;
    if (i < n2) {
        gcf32 gi = (gcf32)in;
        float a = gi[2 * i], b = gi[2 * i + 1];
        ((gu32)out)[i] = pack2(a, b);
    }
}

// ---------- f32 W[K][512] -> bf16 Wt[512][K] (transpose + convert) ----------
__global__ void conv_wt_kernel(const float* __restrict__ W,
                               uint16_t* __restrict__ Wt, int K) {
    int i = blockIdx.x * blockDim.x + threadIdx.x;  // output pair index
    int K2 = K >> 1;
    int total = DIM * K2;
    if (i < total) {
        int n  = i / K2;
        int kp = i - n * K2;
        gcf32 gw = (gcf32)W;
        float a = gw[(2 * kp) * DIM + n];
        float b = gw[(2 * kp + 1) * DIM + n];
        ((gu32)Wt)[i] = pack2(a, b);
    }
}

// ---------- node neighbor mean: (sum nodes[adj] + sum edges[eid]) / (c1+c2) ----------
__global__ __launch_bounds__(256) void node_mean_kernel(
    const uint16_t* __restrict__ nodes, const uint16_t* __restrict__ edges,
    const int* __restrict__ adj, const int* __restrict__ eid,
    uint16_t* __restrict__ out) {
    __shared__ int s_idx[2 * DEG];
    int node = blockIdx.x;
    int tid  = threadIdx.x;
    if (tid < DEG)           s_idx[tid] = ((const AS1 int*)adj)[node * DEG + tid];
    else if (tid < 2 * DEG)  s_idx[tid] = ((const AS1 int*)eid)[node * DEG + (tid - DEG)];
    __syncthreads();
    gcu32 n32 = (gcu32)nodes;
    gcu32 e32 = (gcu32)edges;
    float a0 = 0.f, a1 = 0.f; int cnt = 0;
#pragma unroll
    for (int j = 0; j < DEG; ++j) {
        int idx = s_idx[j];
        if (idx >= 0) {
            uint32_t v = n32[idx * (DIM / 2) + tid];
            a0 += bf2f((uint16_t)v); a1 += bf2f((uint16_t)(v >> 16)); cnt++;
        }
    }
#pragma unroll
    for (int j = 0; j < DEG; ++j) {
        int e = s_idx[DEG + j];
        if (e >= 0) {
            uint32_t v = e32[e * (DIM / 2) + tid];
            a0 += bf2f((uint16_t)v); a1 += bf2f((uint16_t)(v >> 16)); cnt++;
        }
    }
    float inv = 1.0f / (float)cnt;
    ((gu32)out)[node * (DIM / 2) + tid] = pack2(a0 * inv, a1 * inv);
}

// ---------- edge dependency means ----------
__global__ __launch_bounds__(256) void edge_dep_kernel(
    const uint16_t* __restrict__ fw, const uint16_t* __restrict__ bw,
    const int* __restrict__ fidx, const int* __restrict__ bidx,
    uint16_t* __restrict__ fdep, uint16_t* __restrict__ bdep) {
    __shared__ int s_idx[2 * DEP];
    int e   = blockIdx.x;
    int tid = threadIdx.x;
    if (tid < DEP)           s_idx[tid] = ((const AS1 int*)fidx)[e * DEP + tid];
    else if (tid < 2 * DEP)  s_idx[tid] = ((const AS1 int*)bidx)[e * DEP + (tid - DEP)];
    __syncthreads();
    gcu32 f32p = (gcu32)fw;
    gcu32 b32p = (gcu32)bw;
    float f0 = 0.f, f1 = 0.f; int fc = 0;
    float b0 = 0.f, b1 = 0.f; int bc = 0;
#pragma unroll
    for (int j = 0; j < DEP; ++j) {
        int idx = s_idx[j];
        if (idx >= 0) {
            uint32_t v = f32p[idx * (DIM / 2) + tid];
            f0 += bf2f((uint16_t)v); f1 += bf2f((uint16_t)(v >> 16)); fc++;
        }
    }
#pragma unroll
    for (int j = 0; j < DEP; ++j) {
        int idx = s_idx[DEP + j];
        if (idx >= 0) {
            uint32_t v = b32p[idx * (DIM / 2) + tid];
            b0 += bf2f((uint16_t)v); b1 += bf2f((uint16_t)(v >> 16)); bc++;
        }
    }
    float fi = 1.0f / (float)fc, bi = 1.0f / (float)bc;
    ((gu32)fdep)[e * (DIM / 2) + tid] = pack2(f0 * fi, f1 * fi);
    ((gu32)bdep)[e * (DIM / 2) + tid] = pack2(b0 * bi, b1 * bi);
}

// ---------- WMMA GEMM: out[M,512] = concat_K(A0..A_{P-1}) @ W + bias ----------
// Weights pre-transposed: Wt[n][k], k contiguous. Double-buffered LDS pipeline.
#define BLK_M 128
#define BLK_N 64
#define BLK_K 32
#define SA 40   // LDS row stride (elems) for A [128][32]; 80B rows -> 16B aligned
#define SB 40   // LDS row stride (elems) for B stored [64 cols][32 k]

template <int P>
__global__ __launch_bounds__(256, 2) void gemm_wmma_kernel(
    const uint16_t* __restrict__ A0, const uint16_t* __restrict__ A1,
    const uint16_t* __restrict__ A2, const uint16_t* __restrict__ Wt,
    const float* __restrict__ bias, void* __restrict__ out,
    int outF32, int doRelu) {
    constexpr int KTOT   = P * DIM;
    constexpr int NSTEPS = P * (DIM / BLK_K);
    __shared__ __align__(16) uint16_t As[2][BLK_M * SA];
    __shared__ __align__(16) uint16_t Bs[2][BLK_N * SB];

    gcu16 gA0 = (gcu16)A0;
    gcu16 gA1 = (gcu16)A1;
    gcu16 gA2 = (gcu16)A2;
    gcu16 gW  = (gcu16)Wt;

    const int m0   = blockIdx.y * BLK_M;
    const int n0   = blockIdx.x * BLK_N;
    const int tid  = threadIdx.x;
    const int lane = tid & 31;
    const int wave = tid >> 5;
    const int wm   = (wave & 3) * 32;
    const int wn   = (wave >> 2) * 32;

    // staging assignment: A: 2 threads/row, 32B each; B: 4 threads/col, 16B each
    const int rowA = tid >> 1, halfA = tid & 1;
    const int colB = tid >> 2, chunk = tid & 3;
    const int aBase = (m0 + rowA) * DIM + halfA * 16;   // + kIn per step
    const int bBase = (n0 + colB) * KTOT + chunk * 8;   // + kt*32 per step
    const int aDst  = rowA * SA + halfA * 16;
    const int bDst  = colB * SB + chunk * 8;

    u32x4 ra0, ra1, rb;
    auto fetch = [&](int kt) {
        gcu16 Ap = gA0;
        if (kt >= 16) Ap = gA1;
        if (P > 2 && kt >= 32) Ap = gA2;
        int kIn = (kt & 15) * BLK_K;
        gcu16 ap = Ap + (aBase + kIn);
        ra0 = *(gcv4)(ap);
        ra1 = *(gcv4)(ap + 8);
        rb  = *(gcv4)(gW + (bBase + kt * BLK_K));
    };
    auto stage = [&](int buf) {
        *(u32x4*)(&As[buf][aDst])     = ra0;
        *(u32x4*)(&As[buf][aDst + 8]) = ra1;
        *(u32x4*)(&Bs[buf][bDst])     = rb;
    };

    v8f acc[2][2];
    v8f zero = {0.f, 0.f, 0.f, 0.f, 0.f, 0.f, 0.f, 0.f};
    acc[0][0] = zero; acc[0][1] = zero; acc[1][0] = zero; acc[1][1] = zero;

    fetch(0);
    stage(0);

    const int kbA = (lane >> 4) * 8;    // A: half-lanes interleave K in 8-chunks
    const int kbB = (lane >> 4) * 8;    // B: half-lanes interleave K in 8-chunks
    const int ml  = lane & 15;

    int buf = 0;
    for (int kt = 0; kt < NSTEPS; ++kt) {
        if (kt + 1 < NSTEPS) fetch(kt + 1);
        __syncthreads();

        union U { u32x4 q[2]; v16bf v; };
        U a[2], b[2];
#pragma unroll
        for (int i = 0; i < 2; ++i) {
            const uint16_t* ar = &As[buf][(wm + i * 16 + ml) * SA];
            a[i].q[0] = *(const u32x4*)(ar + kbA);
            a[i].q[1] = *(const u32x4*)(ar + 16 + kbA);
            const uint16_t* br = &Bs[buf][(wn + i * 16 + ml) * SB + kbB];
            b[i].q[0] = *(const u32x4*)(br);
            b[i].q[1] = *(const u32x4*)(br + 16);
        }
#pragma unroll
        for (int i = 0; i < 2; ++i)
#pragma unroll
            for (int j = 0; j < 2; ++j)
                acc[i][j] = __builtin_amdgcn_wmma_f32_16x16x32_bf16(
                    false, a[i].v, false, b[j].v, (short)0, acc[i][j], false, false);

        if (kt + 1 < NSTEPS) stage(buf ^ 1);
        buf ^= 1;
    }

    // epilogue: C layout — VGPR r: M = r + 8*(lane>=16), N = lane&15
    gcf32 gb = (gcf32)bias;
    gf32  of = (gf32)out;
    gu16  ob = (gu16)out;
    const int rOff = (lane >> 4) * 8;
#pragma unroll
    for (int i = 0; i < 2; ++i) {
#pragma unroll
        for (int j = 0; j < 2; ++j) {
            int col  = n0 + wn + j * 16 + ml;
            float bv = gb[col];
#pragma unroll
            for (int r = 0; r < 8; ++r) {
                int row = m0 + wm + i * 16 + rOff + r;
                float v = acc[i][j][r] + bv;
                if (doRelu) v = fmaxf(v, 0.f);
                if (outF32) of[(size_t)row * DIM + col] = v;
                else        ob[(size_t)row * DIM + col] = f2bf(v);
            }
        }
    }
}

// ---------- host orchestration ----------
extern "C" void kernel_launch(void* const* d_in, const int* in_sizes, int n_in,
                              void* d_out, int out_size, void* d_ws, size_t ws_size,
                              hipStream_t stream) {
    (void)in_sizes; (void)n_in; (void)out_size; (void)ws_size;
    const float* fw_in   = (const float*)d_in[0];
    const float* bw_in   = (const float*)d_in[1];
    const float* edge_in = (const float*)d_in[2];
    const int*   fw_adj  = (const int*)d_in[3];
    const int*   bw_adj  = (const int*)d_in[4];
    const int*   fw_eid  = (const int*)d_in[5];
    const int*   bw_eid  = (const int*)d_in[6];
    const int*   fw_dep  = (const int*)d_in[7];
    const int*   bw_dep  = (const int*)d_in[8];
    const float* Wfc = (const float*)d_in[9];
    const float* bfc = (const float*)d_in[10];
    const float* Wbc = (const float*)d_in[11];
    const float* bbc = (const float*)d_in[12];
    const float* We  = (const float*)d_in[13];
    const float* be  = (const float*)d_in[14];

    const size_t ND = (size_t)N_NODES * DIM;
    const size_t ED = (size_t)N_EDGES * DIM;

    uint8_t* p = (uint8_t*)d_ws;
    auto alloc = [&](size_t elems) {
        uint16_t* r = (uint16_t*)p;
        p += ((elems * 2 + 255) & ~(size_t)255);
        return r;
    };
    uint16_t* fwA   = alloc(ND);
    uint16_t* bwA   = alloc(ND);
    uint16_t* fhA   = alloc(ND);
    uint16_t* bhA   = alloc(ND);
    uint16_t* fmean = alloc(ND);
    uint16_t* bmean = alloc(ND);
    uint16_t* edgeA = alloc(ED);
    uint16_t* edgeB = alloc(ED);
    uint16_t* fdep  = alloc(ED);
    uint16_t* bdep  = alloc(ED);
    uint16_t* wfcT  = alloc((size_t)2 * DIM * DIM);  // [512][1024]
    uint16_t* wbcT  = alloc((size_t)2 * DIM * DIM);
    uint16_t* wedT  = alloc((size_t)3 * DIM * DIM);  // [512][1536]

    auto conv = [&](const float* src, uint16_t* dst, size_t elems) {
        int n2 = (int)(elems / 2);
        conv_bf16_kernel<<<(n2 + 255) / 256, 256, 0, stream>>>(src, dst, n2);
    };
    auto convT = [&](const float* src, uint16_t* dst, int K) {
        int total = DIM * (K / 2);
        conv_wt_kernel<<<(total + 255) / 256, 256, 0, stream>>>(src, dst, K);
    };
    conv(fw_in, fwA, ND);
    conv(bw_in, bwA, ND);
    conv(edge_in, edgeA, ED);
    convT(Wfc, wfcT, 2 * DIM);
    convT(Wbc, wbcT, 2 * DIM);
    convT(We,  wedT, 3 * DIM);

    dim3 gNode(DIM / BLK_N, N_NODES / BLK_M);   // (8, 64)
    dim3 gEdge(DIM / BLK_N, N_EDGES / BLK_M);   // (8, 256)
    float* out_fw = (float*)d_out;
    float* out_bw = (float*)d_out + ND;

    for (int k = 0; k < HOPS; ++k) {
        const bool last = (k == HOPS - 1);
        node_mean_kernel<<<N_NODES, 256, 0, stream>>>(fwA, edgeA, fw_adj, fw_eid, fmean);
        node_mean_kernel<<<N_NODES, 256, 0, stream>>>(bwA, edgeA, bw_adj, bw_eid, bmean);

        gemm_wmma_kernel<2><<<gNode, 256, 0, stream>>>(
            fwA, fmean, (const uint16_t*)nullptr, wfcT, bfc,
            last ? (void*)out_fw : (void*)fhA, last ? 1 : 0, last ? 0 : 1);
        gemm_wmma_kernel<2><<<gNode, 256, 0, stream>>>(
            bwA, bmean, (const uint16_t*)nullptr, wbcT, bbc,
            last ? (void*)out_bw : (void*)bhA, last ? 1 : 0, last ? 0 : 1);

        uint16_t* t;
        t = fwA; fwA = fhA; fhA = t;
        t = bwA; bwA = bhA; bhA = t;

        if (!last) {
            edge_dep_kernel<<<N_EDGES, 256, 0, stream>>>(fwA, bwA, fw_dep, bw_dep, fdep, bdep);
            gemm_wmma_kernel<3><<<gEdge, 256, 0, stream>>>(
                edgeA, fdep, bdep, wedT, be, (void*)edgeB, 0, 1);
            t = edgeA; edgeA = edgeB; edgeB = t;
        }
    }
}
